// SelfAttention_37795712205022
// MI455X (gfx1250) — matrix operand order repaired and checked by reading, not compiled, not run on hardware
//
#include <hip/hip_runtime.h>
#include <hip/hip_fp16.h>

typedef __attribute__((ext_vector_type(16))) _Float16 v16h;
typedef __attribute__((ext_vector_type(8)))  _Float16 v8h;
typedef __attribute__((ext_vector_type(4)))  _Float16 v4h;
typedef __attribute__((ext_vector_type(8)))  float    v8f;

#define HH   16      // heads
#define DH   64      // head dim
#define LSEQ 2048
#define DMOD 1024
#define BB   4
#define NEGV (-1.0e12f)

// Load 2x16B of f16 into one 32B WMMA fragment half-pair.
static __device__ inline v16h ld16h(const _Float16* p0, const _Float16* p1) {
    v8h a = *(const v8h*)p0;
    v8h b = *(const v8h*)p1;
    return __builtin_shufflevector(a, b, 0, 1, 2, 3, 4, 5, 6, 7,
                                         8, 9, 10, 11, 12, 13, 14, 15);
}

// ---------------------------------------------------------------------------
// Tensor Data Mover: 2-D tile (rows x row_len f16, row stride in elements)
// from global to LDS.  D# packed per CDNA5 ISA.
// ---------------------------------------------------------------------------
#if __has_builtin(__builtin_amdgcn_tensor_load_to_lds)
#define USE_TDM 1
typedef unsigned int u32x4 __attribute__((ext_vector_type(4)));
typedef int          i32x4 __attribute__((ext_vector_type(4)));
typedef int          i32x8 __attribute__((ext_vector_type(8)));

static __device__ inline void tdm_load_tile_f16(const void* gptr, unsigned int lds_off,
                                                unsigned int row_len, unsigned int rows,
                                                unsigned long long stride_elems)
{
    const unsigned long long ga = (unsigned long long)(uintptr_t)gptr;
    u32x4 g0;
    g0[0] = 1u;                                              // count=1, no gather
    g0[1] = lds_off;                                         // lds_addr (bytes)
    g0[2] = (unsigned int)(ga & 0xFFFFFFFFu);                // global_addr[31:0]
    g0[3] = (unsigned int)((ga >> 32) & 0x1FFFFFFu)          // global_addr[56:32]
          | (2u << 30);                                      // type = 2 ("image")
    i32x8 g1;
    g1[0] = (int)(1u << 16);                                 // data_size = 2 bytes
    g1[1] = (int)((row_len & 0xFFFFu) << 16);                // tensor_dim0[15:0]
    g1[2] = (int)((row_len >> 16) | ((rows & 0xFFFFu) << 16));   // td0 hi | td1 lo
    g1[3] = (int)((rows >> 16) | ((row_len & 0xFFFFu) << 16));   // td1 hi | tile_dim0
    g1[4] = (int)(rows & 0xFFFFu);                           // tile_dim1 (tile_dim2=0)
    g1[5] = (int)(unsigned int)(stride_elems & 0xFFFFFFFFu); // tensor_dim0_stride lo
    g1[6] = (int)(unsigned int)((stride_elems >> 32) & 0xFFFFu); // stride hi
    g1[7] = 0;                                               // tensor_dim1_stride = 0
    i32x4 z4 = {0, 0, 0, 0};                                 // 2-D: groups 2/3 unused
#if __clang_major__ >= 23
    i32x8 z8 = {0, 0, 0, 0, 0, 0, 0, 0};
    __builtin_amdgcn_tensor_load_to_lds(g0, g1, z4, z4, z8, 0);
#else
    __builtin_amdgcn_tensor_load_to_lds(g0, g1, z4, z4, 0);
#endif
}

static __device__ inline void wait_tensorcnt0() {
#if __has_builtin(__builtin_amdgcn_s_wait_tensorcnt)
    __builtin_amdgcn_s_wait_tensorcnt(0);
#else
    asm volatile("s_wait_tensorcnt 0" ::: "memory");
#endif
}
#endif

// ---------------------------------------------------------------------------
// fp32 -> f16 bulk convert (the three weight matrices, 1M elems each)
// ---------------------------------------------------------------------------
__global__ __launch_bounds__(256)
void cvt_f32_to_f16(const float* __restrict__ in, _Float16* __restrict__ outp, int n)
{
    const int i = (blockIdx.x * 256 + threadIdx.x) * 4;
    if (i + 3 < n) {
        const float4 f = *(const float4*)(in + i);
        v4h o;
        o[0] = (_Float16)f.x; o[1] = (_Float16)f.y;
        o[2] = (_Float16)f.z; o[3] = (_Float16)f.w;
        *(v4h*)(outp + i) = o;
    }
}

// ---------------------------------------------------------------------------
// Projection GEMM: Y[m,n] = sum_d X[m,d] * Wh[n,d] + b[n]
// X fp32 (converted inline), Wh pre-converted f16 (b128 loads).
// Block = 8 waves; wave = 16x64 tile; K-loop 32 steps of K=32.
// Output f16: [B,H,L,DH] (Q,K) or [B,H,DH,L] (V transposed).
// ---------------------------------------------------------------------------
__global__ __launch_bounds__(256)
void qkv_proj_kernel(const float* __restrict__ x, const _Float16* __restrict__ wh,
                     const float* __restrict__ bias, _Float16* __restrict__ out,
                     int transposed)
{
    const int lane = threadIdx.x & 31;
    const int wave = threadIdx.x >> 5;
    const int lo   = lane & 15;
    const int hi   = lane >> 4;

    const int m0 = blockIdx.x * 128 + wave * 16;
    const int n0 = blockIdx.y * 64;

    v8f acc[4] = {};
    const float* xrow = x + (size_t)(m0 + lo) * DMOD;

    for (int kb = 0; kb < DMOD; kb += 32) {
        v16h a;
        {
            const float* p0 = xrow + kb + hi * 8;
            #pragma unroll
            for (int i = 0; i < 8; ++i) a[i]     = (_Float16)p0[i];
            #pragma unroll
            for (int i = 0; i < 8; ++i) a[8 + i] = (_Float16)p0[16 + i];
        }
        #pragma unroll
        for (int s = 0; s < 4; ++s) {
            const _Float16* wp = wh + (size_t)(n0 + s * 16 + lo) * DMOD + kb + hi * 8;
            const v16h bf = ld16h(wp, wp + 16);
            acc[s] = __builtin_amdgcn_wmma_f32_16x16x32_f16(
                         false, a, false, bf, (short)0, acc[s], false, false);
        }
    }

    #pragma unroll
    for (int s = 0; s < 4; ++s) {
        const int n  = n0 + s * 16 + lo;
        const float bv = bias[n];
        const int hh = n >> 6;
        const int dh = n & 63;
        #pragma unroll
        for (int v = 0; v < 8; ++v) {
            const int m    = m0 + v + hi * 8;
            const int bidx = m >> 11;
            const int l    = m & (LSEQ - 1);
            const float y  = acc[s][v] + bv;
            size_t off;
            if (transposed)
                off = ((size_t)(bidx * HH + hh) * DH + dh) * LSEQ + l;
            else
                off = ((size_t)(bidx * HH + hh) * LSEQ + l) * DH + dh;
            out[off] = (_Float16)y;
        }
    }
}

// ---------------------------------------------------------------------------
// Flash attention, block-cooperative k-loop:
//  - K tile (32 x 64) and V tile (64 x 32) staged into LDS once per k-step
//    via TENSOR_LOAD_TO_LDS (wave 0 issues; s_wait_tensorcnt; barrier).
//  - each wave owns a 16-row q tile; online softmax with causal+padding masks.
// ---------------------------------------------------------------------------
__global__ __launch_bounds__(256)
void flash_attn_kernel(const _Float16* __restrict__ Qh,
                       const _Float16* __restrict__ Kh,
                       const _Float16* __restrict__ Vt,
                       const int* __restrict__ q_len,
                       const int* __restrict__ v_len,
                       float* __restrict__ out)
{
    __shared__ _Float16 lds_k[32 * 64];       // K tile: 32 rows x 64 d
    __shared__ _Float16 lds_v[64 * 32];       // V tile: 64 d-rows x 32 k
    __shared__ _Float16 lds_p[8 * 16 * 32];   // per-wave 16x32 P staging

    const int lane = threadIdx.x & 31;
    const int wave = threadIdx.x >> 5;
    const int lo   = lane & 15;
    const int hi   = lane >> 4;

    const int qtiles = LSEQ / 128;
    const int qt = blockIdx.x % qtiles;
    const int bh = blockIdx.x / qtiles;
    const int b  = bh / HH;
    const int h  = bh % HH;

    const int qb0  = qt * 128;
    const int q0   = qb0 + wave * 16;
    const int vlen = v_len[b];
    const int qlen = q_len[b];

    const _Float16* qbase = Qh + (size_t)bh * LSEQ * DH;
    const _Float16* kbase = Kh + (size_t)bh * LSEQ * DH;
    const _Float16* vbase = Vt + (size_t)bh * DH * LSEQ;

    // persistent Q A-fragments, d = [0,32) and [32,64)
    v16h qa[2];
    {
        const _Float16* qr = qbase + (size_t)(q0 + lo) * DH;
        #pragma unroll
        for (int f = 0; f < 2; ++f) {
            const _Float16* p0 = qr + f * 32 + hi * 8;
            qa[f] = ld16h(p0, p0 + 16);
        }
    }

    v8f oacc[4] = {};
    float rm[8], rl[8];
    #pragma unroll
    for (int v = 0; v < 8; ++v) { rm[v] = -1.0e30f; rl[v] = 0.0f; }

    const int kend_w   = min(q0 + 16, vlen);    // this wave's last useful column
    const int kend_blk = min(qb0 + 128, vlen);  // block-uniform loop bound
    _Float16* myp = lds_p + wave * (16 * 32);

    for (int kb = 0; kb < kend_blk; kb += 32) {
        // ---- stage K tile (32x64, contiguous) and V tile (64x32, stride L) ----
#ifdef USE_TDM
        if (wave == 0) {
            tdm_load_tile_f16(kbase + (size_t)kb * DH,
                              (unsigned int)(uintptr_t)(void*)lds_k,
                              /*row_len=*/DH, /*rows=*/32, /*stride=*/DH);
            tdm_load_tile_f16(vbase + kb,
                              (unsigned int)(uintptr_t)(void*)lds_v,
                              /*row_len=*/32, /*rows=*/DH, /*stride=*/LSEQ);
            wait_tensorcnt0();
        }
        __syncthreads();
#else
        {   // cooperative fallback: 256 threads x 16B each per tile
            const int tid = threadIdx.x;
            *((v8h*)lds_k + tid) = *((const v8h*)(kbase + (size_t)kb * DH) + tid);
            const int vr = tid >> 2, vc = (tid & 3) * 8;
            *(v8h*)(lds_v + vr * 32 + vc) =
                *(const v8h*)(vbase + (size_t)vr * LSEQ + kb + vc);
        }
        __syncthreads();
#endif

        if (kb < kend_w) {
            // ----- S = Q K^T (two 16-col tiles, B-fragments from LDS) -----
            v8f s[2] = {};
            #pragma unroll
            for (int t = 0; t < 2; ++t) {
                const _Float16* kr = lds_k + (t * 16 + lo) * DH;
                #pragma unroll
                for (int f = 0; f < 2; ++f) {
                    const _Float16* p = kr + f * 32 + hi * 8;
                    const v16h bf = ld16h(p, p + 16);
                    s[t] = __builtin_amdgcn_wmma_f32_16x16x32_f16(
                               false, qa[f], false, bf, (short)0, s[t], false, false);
                }
            }

            // ----- masks + online softmax (16-lane shfl reductions) -----
            const int kc0 = kb + lo;
            const int kc1 = kb + 16 + lo;
            #pragma unroll
            for (int v = 0; v < 8; ++v) {
                const int qrow = q0 + v + hi * 8;
                float s0 = s[0][v] * 0.125f;               // 1/sqrt(64)
                float s1 = s[1][v] * 0.125f;
                if (kc0 > qrow || kc0 >= vlen) s0 = NEGV;
                if (kc1 > qrow || kc1 >= vlen) s1 = NEGV;

                float mx = fmaxf(s0, s1);
                #pragma unroll
                for (int d = 8; d >= 1; d >>= 1) mx = fmaxf(mx, __shfl_xor(mx, d, 32));
                mx = fmaxf(mx, rm[v]);

                const float p0 = __expf(s0 - mx);
                const float p1 = __expf(s1 - mx);
                float ps = p0 + p1;
                #pragma unroll
                for (int d = 8; d >= 1; d >>= 1) ps += __shfl_xor(ps, d, 32);

                const float alpha = __expf(rm[v] - mx);
                rl[v] = rl[v] * alpha + ps;
                rm[v] = mx;
                #pragma unroll
                for (int sn = 0; sn < 4; ++sn) oacc[sn][v] *= alpha;

                myp[(v + hi * 8) * 32 + lo]      = (_Float16)p0;
                myp[(v + hi * 8) * 32 + 16 + lo] = (_Float16)p1;
            }

            asm volatile("s_wait_dscnt 0" ::: "memory");   // wave-local LDS RAW

            // reload P as A-fragment (16x32 over k)
            v16h pa;
            {
                const _Float16* pr = myp + lo * 32 + hi * 8;
                pa = ld16h(pr, pr + 16);
            }

            // ----- O += P V  (V tile from LDS; lane = d column) -----
            #pragma unroll
            for (int sn = 0; sn < 4; ++sn) {
                const _Float16* vr = lds_v + (sn * 16 + lo) * 32 + hi * 8;
                const v16h vf = ld16h(vr, vr + 16);
                oacc[sn] = __builtin_amdgcn_wmma_f32_16x16x32_f16(
                               false, pa, false, vf, (short)0, oacc[sn], false, false);
            }
        }
        __syncthreads();   // protect LDS tiles before next TDM overwrite
    }

    // ----- finalize: O / l, query-padding mask, write [B,L,H*DH] f32 -----
    #pragma unroll
    for (int v = 0; v < 8; ++v) {
        const int qrow = q0 + v + hi * 8;
        const float inv = (qrow < qlen) ? (1.0f / rl[v]) : 0.0f;
        float* orow = out + (size_t)(b * LSEQ + qrow) * (HH * DH) + h * DH;
        #pragma unroll
        for (int sn = 0; sn < 4; ++sn)
            orow[sn * 16 + lo] = oacc[sn][v] * inv;
    }
}

// ---------------------------------------------------------------------------
extern "C" void kernel_launch(void* const* d_in, const int* in_sizes, int n_in,
                              void* d_out, int out_size, void* d_ws, size_t ws_size,
                              hipStream_t stream)
{
    const float* q_seq = (const float*)d_in[0];
    const float* k_seq = (const float*)d_in[1];
    const float* v_seq = (const float*)d_in[2];
    const int*   q_len = (const int*)d_in[3];
    const int*   v_len = (const int*)d_in[4];
    const float* WQ_w  = (const float*)d_in[5];
    const float* WQ_b  = (const float*)d_in[6];
    const float* WK_w  = (const float*)d_in[7];
    const float* WK_b  = (const float*)d_in[8];
    const float* WV_w  = (const float*)d_in[9];
    const float* WV_b  = (const float*)d_in[10];

    const size_t qkv_elems = (size_t)BB * HH * LSEQ * DH;   // 8.4M f16 each
    const size_t w_elems   = (size_t)DMOD * DMOD;           // 1M f16 each

    _Float16* Qh  = (_Float16*)d_ws;
    _Float16* Kh  = Qh + qkv_elems;
    _Float16* Vt  = Kh + qkv_elems;
    _Float16* WQh = Vt + qkv_elems;
    _Float16* WKh = WQh + w_elems;
    _Float16* WVh = WKh + w_elems;

    // 1) weights -> f16
    const int cvt_blocks = (int)(w_elems / (256 * 4));
    cvt_f32_to_f16<<<cvt_blocks, 256, 0, stream>>>(WQ_w, WQh, (int)w_elems);
    cvt_f32_to_f16<<<cvt_blocks, 256, 0, stream>>>(WK_w, WKh, (int)w_elems);
    cvt_f32_to_f16<<<cvt_blocks, 256, 0, stream>>>(WV_w, WVh, (int)w_elems);

    // 2) projections (f16 WMMA, f32 accumulate)
    dim3 gp(64, 16, 1);   // (8192/128) x (1024/64)
    qkv_proj_kernel<<<gp, 256, 0, stream>>>(q_seq, WQh, WQ_b, Qh, 0);
    qkv_proj_kernel<<<gp, 256, 0, stream>>>(k_seq, WKh, WK_b, Kh, 0);
    qkv_proj_kernel<<<gp, 256, 0, stream>>>(v_seq, WVh, WV_b, Vt, 1);

    // 3) flash attention with TDM-staged K/V tiles
    flash_attn_kernel<<<dim3(BB * HH * (LSEQ / 128)), 256, 0, stream>>>(
        Qh, Kh, Vt, q_len, v_len, (float*)d_out);
}
